// MLPMixedAdapterBlock_17102559772721
// MI455X (gfx1250) — matrix operand order repaired and checked by reading, not compiled, not run on hardware
//
#include <hip/hip_runtime.h>

typedef unsigned short u16;
typedef __attribute__((ext_vector_type(8)))  unsigned short v8u;
typedef __attribute__((ext_vector_type(8)))  __bf16 v8bf;
typedef __attribute__((ext_vector_type(16))) __bf16 v16bf;
typedef __attribute__((ext_vector_type(8)))  float v8f;
typedef int v4i __attribute__((vector_size(16)));   // GCC vector: matches builtin proto

#define S_   1024
#define D_   768
#define H_   12
#define HD_  64
#define FF_  3072
#define R_   16
#define NB_  768
#define M_   8192   /* B*S */

// Async global->LDS path (CDNA5): guarded so a name/signature mismatch falls back
// to synchronous staging instead of breaking the build.
#if defined(__gfx1250__) && __has_builtin(__builtin_amdgcn_global_load_async_to_lds_b128) && \
    __has_builtin(__builtin_amdgcn_s_wait_asynccnt)
#define ASYNC_LDS 1
#else
#define ASYNC_LDS 0
#endif

// ---------------- device helpers ----------------
__device__ __forceinline__ u16 f2bfu(float f) {
  unsigned u = __builtin_bit_cast(unsigned, f);
  u += 0x7FFFu + ((u >> 16) & 1u);      // round-to-nearest-even
  return (u16)(u >> 16);
}
__device__ __forceinline__ float bfu2f(u16 h) {
  return __builtin_bit_cast(float, ((unsigned)h) << 16);
}
__device__ __forceinline__ v16bf catbf(v8bf lo, v8bf hi) {
  return __builtin_shufflevector(lo, hi, 0,1,2,3,4,5,6,7,8,9,10,11,12,13,14,15);
}
__device__ __forceinline__ v8f wmma_bf16(v16bf a, v16bf b, v8f c) {
  return __builtin_amdgcn_wmma_f32_16x16x32_bf16(false, a, false, b, (short)0, c,
                                                 false, false);
}
__device__ __forceinline__ v8f vzero8() {
  v8f z = {0.f,0.f,0.f,0.f,0.f,0.f,0.f,0.f};
  return z;
}

#if ASYNC_LDS
// 16B per lane, wave32 -> 512B per instruction; tracked by ASYNCcnt.
__device__ __forceinline__ void async_b128(const void* g, void* l) {
  __builtin_amdgcn_global_load_async_to_lds_b128(
      (__attribute__((address_space(1))) v4i*)(unsigned long long)g,
      (__attribute__((address_space(3))) v4i*)(unsigned)(unsigned long long)l,
      0, 0);
}
#endif

// ---------------- fp32 -> bf16 convert ----------------
__global__ __launch_bounds__(256) void cvt_bf16_kernel(const float* __restrict__ in,
                                                       u16* __restrict__ out, int n) {
  int i = blockIdx.x * 256 + threadIdx.x;
  if (i < n) out[i] = f2bfu(in[i]);
}

// ---------------- LayerNorm (f32 in -> bf16 out), row = D_ = 768 ----------------
__global__ __launch_bounds__(256) void ln_bf16_kernel(const float* __restrict__ x,
                                                      const float* __restrict__ g,
                                                      const float* __restrict__ bb,
                                                      u16* __restrict__ out) {
  int row = blockIdx.x;
  int tid = threadIdx.x;
  const float* xr = x + (size_t)row * D_;
  float v0 = xr[tid], v1 = xr[tid + 256], v2 = xr[tid + 512];
  __shared__ float rs[256], rq[256];
  rs[tid] = v0 + v1 + v2;
  rq[tid] = v0 * v0 + v1 * v1 + v2 * v2;
  __syncthreads();
  for (int off = 128; off > 0; off >>= 1) {
    if (tid < off) { rs[tid] += rs[tid + off]; rq[tid] += rq[tid + off]; }
    __syncthreads();
  }
  float mean = rs[0] * (1.f / D_);
  float var  = rq[0] * (1.f / D_) - mean * mean;
  float inv  = rsqrtf(var + 1e-6f);
  u16* orow = out + (size_t)row * D_;
  orow[tid]       = f2bfu((v0 - mean) * inv * g[tid]       + bb[tid]);
  orow[tid + 256] = f2bfu((v1 - mean) * inv * g[tid + 256] + bb[tid + 256]);
  orow[tid + 512] = f2bfu((v2 - mean) * inv * g[tid + 512] + bb[tid + 512]);
}

// ---------------- tiled WMMA GEMM: C[M,N] = A[M,K] @ W[N,K]^T + bias ----------------
// mode 0: f32 out (+res if res!=null)   mode 1: bf16 out   mode 2: bf16 head-layout*scale
// Double-buffered LDS; async global->LDS prefetch of tile k+1 overlaps WMMA on tile k.
__global__ __launch_bounds__(256) void gemm_bf16_kernel(const u16* __restrict__ A,
                                                        const u16* __restrict__ W,
                                                        const float* __restrict__ bias,
                                                        const float* __restrict__ res,
                                                        float* __restrict__ outf,
                                                        u16* __restrict__ outb,
                                                        int M, int N, int K,
                                                        int mode, float scale) {
  __shared__ alignas(16) u16 At[2][128 * 32];
  __shared__ alignas(16) u16 Bt[2][128 * 32];
  int tid = threadIdx.x, lane = tid & 31, wave = tid >> 5;
  int wm = wave & 3, wn = wave >> 2;
  int mBase = blockIdx.y * 128, nBase = blockIdx.x * 128;
  int lr   = lane & 15;
  int klo  = (lane < 16) ? 0 : 8;    // A fragment K-chunk start (halves)
  int koff = (lane < 16) ? 0 : 8;    // B fragment K-chunk start (halves)

  v8f acc[2][4];
#pragma unroll
  for (int mi = 0; mi < 2; ++mi)
#pragma unroll
    for (int ni = 0; ni < 4; ++ni) acc[mi][ni] = vzero8();

  auto compute = [&](int cur) {
    v16bf af[2], bfr[4];
#pragma unroll
    for (int mi = 0; mi < 2; ++mi) {
      int r = wm * 32 + mi * 16 + lr;
      v8bf lo = *(const v8bf*)&At[cur][r * 32 + klo];
      v8bf hi = *(const v8bf*)&At[cur][r * 32 + klo + 16];
      af[mi] = catbf(lo, hi);
    }
#pragma unroll
    for (int ni = 0; ni < 4; ++ni) {
      int r = wn * 64 + ni * 16 + lr;
      v8bf lo = *(const v8bf*)&Bt[cur][r * 32 + koff];
      v8bf hi = *(const v8bf*)&Bt[cur][r * 32 + koff + 16];
      bfr[ni] = catbf(lo, hi);
    }
#pragma unroll
    for (int mi = 0; mi < 2; ++mi)
#pragma unroll
      for (int ni = 0; ni < 4; ++ni)
        acc[mi][ni] = wmma_bf16(af[mi], bfr[ni], acc[mi][ni]);
  };

#if ASYNC_LDS
  // prologue: async-stage tile kb=0 into buffer 0 (4 async instructions / wave)
#pragma unroll
  for (int i = 0; i < 2; ++i) {
    int idx = tid * 2 + i, row = idx >> 2, seg = idx & 3;
    async_b128(&A[(size_t)(mBase + row) * K + seg * 8], &At[0][row * 32 + seg * 8]);
    async_b128(&W[(size_t)(nBase + row) * K + seg * 8], &Bt[0][row * 32 + seg * 8]);
  }
  for (int kb = 0; kb < K; kb += 32) {
    int cur = (kb >> 5) & 1;
    __syncthreads();                       // all waves done reading buf cur^1
    if (kb + 32 < K) {
#pragma unroll
      for (int i = 0; i < 2; ++i) {
        int idx = tid * 2 + i, row = idx >> 2, seg = idx & 3;
        async_b128(&A[(size_t)(mBase + row) * K + kb + 32 + seg * 8],
                   &At[cur ^ 1][row * 32 + seg * 8]);
        async_b128(&W[(size_t)(nBase + row) * K + kb + 32 + seg * 8],
                   &Bt[cur ^ 1][row * 32 + seg * 8]);
      }
      __builtin_amdgcn_s_wait_asynccnt(4); // tile kb done; kb+32 still in flight
    } else {
      __builtin_amdgcn_s_wait_asynccnt(0);
    }
    __syncthreads();                       // every wave's tile-kb loads landed
    compute(cur);
  }
#else
  // fallback: synchronous staging through VGPRs (single buffer)
  for (int kb = 0; kb < K; kb += 32) {
    __syncthreads();
#pragma unroll
    for (int i = 0; i < 2; ++i) {
      int idx = tid * 2 + i, row = idx >> 2, seg = idx & 3;
      *(v8u*)&At[0][row * 32 + seg * 8] =
          *(const v8u*)&A[(size_t)(mBase + row) * K + kb + seg * 8];
      *(v8u*)&Bt[0][row * 32 + seg * 8] =
          *(const v8u*)&W[(size_t)(nBase + row) * K + kb + seg * 8];
    }
    __syncthreads();
    compute(0);
  }
#endif

  int rowAdd = (lane < 16) ? 0 : 8;
#pragma unroll
  for (int mi = 0; mi < 2; ++mi) {
#pragma unroll
    for (int ni = 0; ni < 4; ++ni) {
      int mT = mBase + wm * 32 + mi * 16;
      int col = nBase + wn * 64 + ni * 16 + lr;
      float bcol = bias ? bias[col] : 0.f;
#pragma unroll
      for (int r = 0; r < 8; ++r) {
        int row = mT + rowAdd + r;
        float v = acc[mi][ni][r] + bcol;
        size_t o = (size_t)row * N + col;
        if (mode == 0) {
          outf[o] = res ? (v + res[o]) : v;
        } else if (mode == 1) {
          outb[o] = f2bfu(v);
        } else {  // head layout: [B,H,S,64], row=b*S+s, col=h*64+d
          int b  = row >> 10, s = row & (S_ - 1);
          int hh = col >> 6,  d = col & (HD_ - 1);
          size_t oo = ((((size_t)b * H_ + hh) * S_) + s) * HD_ + d;
          outb[oo] = f2bfu(v * scale);
        }
      }
    }
  }
}

// ---------------- flash attention: per (b,h,128 q-rows) block ----------------
__global__ __launch_bounds__(256) void flash_attn_kernel(const u16* __restrict__ qh,
                                                         const u16* __restrict__ kh,
                                                         const u16* __restrict__ vh,
                                                         const float* __restrict__ mask,
                                                         u16* __restrict__ ao) {
  __shared__ alignas(16) u16 Kl[32 * 64];       // [key][d]
  __shared__ alignas(16) u16 Vt[64 * 32];       // [d][key] (transposed)
  __shared__ alignas(16) u16 Pl[8 * 16 * 32];   // per-wave P strip [16 q][32 key]
  int tid = threadIdx.x, lane = tid & 31, wave = tid >> 5;
  int qb = blockIdx.x & 7;            // S/128 = 8 q blocks
  int bh = blockIdx.x >> 3;           // 0..B*H-1
  int b = bh / H_, h = bh % H_;
  const size_t headOff = (size_t)bh * S_ * HD_;
  int lr = lane & 15;
  int klo  = (lane < 16) ? 0 : 8;
  int koff = (lane < 16) ? 0 : 8;
  int rowAdd = (lane < 16) ? 0 : 8;
  int q0 = qb * 128 + wave * 16;

  // q fragments (16 rows x 64 dims -> two K=32 A fragments), q pre-scaled on store
  v16bf qf[2];
  {
    const u16* qrow = qh + headOff + (size_t)(q0 + lr) * HD_;
#pragma unroll
    for (int dc = 0; dc < 2; ++dc) {
      v8bf lo = *(const v8bf*)&qrow[dc * 32 + klo];
      v8bf hi = *(const v8bf*)&qrow[dc * 32 + klo + 16];
      qf[dc] = catbf(lo, hi);
    }
  }

  v8f acc[4];
#pragma unroll
  for (int dt = 0; dt < 4; ++dt) acc[dt] = vzero8();
  float mrow[8], lrow[8];
#pragma unroll
  for (int r = 0; r < 8; ++r) { mrow[r] = -1e30f; lrow[r] = 0.f; }

  for (int kb = 0; kb < S_; kb += 32) {
    __syncthreads();
    {  // cooperative stage: K chunk row-major, V chunk transposed
      int row = tid >> 3, seg = tid & 7;  // 32 keys x 8 segs of 8 halves
      *(v8u*)&Kl[row * 64 + seg * 8] =
          *(const v8u*)&kh[headOff + (size_t)(kb + row) * HD_ + seg * 8];
      v8u vv = *(const v8u*)&vh[headOff + (size_t)(kb + row) * HD_ + seg * 8];
#pragma unroll
      for (int j = 0; j < 8; ++j) Vt[(seg * 8 + j) * 32 + row] = vv[j];
    }
    __syncthreads();

    // scores: s[t] = Q(16x64) @ K_chunk(64 x 16keys), two key tiles
    v8f s[2];
#pragma unroll
    for (int t = 0; t < 2; ++t) {
      s[t] = vzero8();
#pragma unroll
      for (int dc = 0; dc < 2; ++dc) {
        const u16* krow = &Kl[(t * 16 + lr) * 64 + dc * 32];
        v8bf lo = *(const v8bf*)&krow[koff];
        v8bf hi = *(const v8bf*)&krow[koff + 16];
        s[t] = wmma_bf16(qf[dc], catbf(lo, hi), s[t]);
      }
    }

    // online softmax update (rows live in 16-lane halves; VGPR r = row r / r+8)
#pragma unroll
    for (int r = 0; r < 8; ++r) {
      int rowg = q0 + rowAdd + r;
      float s0 = s[0][r] + mask[(size_t)rowg * S_ + kb + lr];
      float s1 = s[1][r] + mask[(size_t)rowg * S_ + kb + 16 + lr];
      float mx = fmaxf(s0, s1);
      mx = fmaxf(mx, __shfl_xor(mx, 1));
      mx = fmaxf(mx, __shfl_xor(mx, 2));
      mx = fmaxf(mx, __shfl_xor(mx, 4));
      mx = fmaxf(mx, __shfl_xor(mx, 8));
      float mnew = fmaxf(mrow[r], mx);
      float corr = __expf(mrow[r] - mnew);
      float p0 = __expf(s0 - mnew);
      float p1 = __expf(s1 - mnew);
      float rsum = p0 + p1;
      rsum += __shfl_xor(rsum, 1);
      rsum += __shfl_xor(rsum, 2);
      rsum += __shfl_xor(rsum, 4);
      rsum += __shfl_xor(rsum, 8);
      lrow[r] = lrow[r] * corr + rsum;
      mrow[r] = mnew;
#pragma unroll
      for (int dt = 0; dt < 4; ++dt) acc[dt][r] *= corr;
      int prow = rowAdd + r;
      Pl[(wave * 16 + prow) * 32 + lr]      = f2bfu(p0);
      Pl[(wave * 16 + prow) * 32 + 16 + lr] = f2bfu(p1);
    }

    // P fragment (16 q x 32 keys)
    v16bf pf;
    {
      const u16* prow = &Pl[(wave * 16 + lr) * 32];
      v8bf lo = *(const v8bf*)&prow[klo];
      v8bf hi = *(const v8bf*)&prow[klo + 16];
      pf = catbf(lo, hi);
    }
    // out(16 x 64) += P @ V_chunk  (B fragment from transposed V)
#pragma unroll
    for (int dt = 0; dt < 4; ++dt) {
      const u16* vr = &Vt[(dt * 16 + lr) * 32];
      v8bf lo = *(const v8bf*)&vr[koff];
      v8bf hi = *(const v8bf*)&vr[koff + 16];
      acc[dt] = wmma_bf16(pf, catbf(lo, hi), acc[dt]);
    }
  }

  // epilogue: merge heads into [B*S, D] bf16
#pragma unroll
  for (int r = 0; r < 8; ++r) {
    float inv = 1.0f / lrow[r];
    int rowg = q0 + rowAdd + r;
    size_t base = ((size_t)b * S_ + rowg) * D_ + h * HD_;
#pragma unroll
    for (int dt = 0; dt < 4; ++dt)
      ao[base + dt * 16 + lr] = f2bfu(acc[dt][r] * inv);
  }
}

// ---------------- adapter down: t[m,r] = (in[m,:] @ A)[r] * (expert? d[r] : 1) -------
__global__ __launch_bounds__(256) void adapter_down_kernel(const u16* __restrict__ in,
                                                           const float* __restrict__ Aback,
                                                           const float* __restrict__ Aexp,
                                                           const float* __restrict__ dexp,
                                                           float* __restrict__ t, int K) {
  int row = blockIdx.x;
  int s = row & (S_ - 1);
  bool expert = (s >= NB_);
  const float* A = expert ? Aexp : Aback;
  int r  = threadIdx.x >> 4;
  int ln = threadIdx.x & 15;
  const u16* inr = in + (size_t)row * K;
  float acc = 0.f;
  for (int k = ln; k < K; k += 16) acc += bfu2f(inr[k]) * A[k * R_ + r];
  acc += __shfl_xor(acc, 1);
  acc += __shfl_xor(acc, 2);
  acc += __shfl_xor(acc, 4);
  acc += __shfl_xor(acc, 8);
  if (ln == 0) t[(size_t)row * R_ + r] = expert ? acc * dexp[r] : acc;
}

// ---------------- FC1 post: hid + adapter-up (+expert col scale), GELU, -> bf16 ------
__global__ __launch_bounds__(256) void fc1_post_kernel(const u16* __restrict__ hid,
                                                       const float* __restrict__ t1,
                                                       const float* __restrict__ Bback,
                                                       const float* __restrict__ Bexp,
                                                       const float* __restrict__ expb,
                                                       u16* __restrict__ out) {
  int idx = blockIdx.x * 256 + threadIdx.x;     // M_*FF_ elements
  int m = idx / FF_, n = idx % FF_;
  int s = m & (S_ - 1);
  bool expert = (s >= NB_);
  const float* Bm = expert ? Bexp : Bback;
  const float* tr = t1 + (size_t)m * R_;
  float add = 0.f;
#pragma unroll
  for (int r = 0; r < R_; ++r) add += tr[r] * Bm[r * FF_ + n];
  if (expert) add *= expb[n];
  float v = bfu2f(hid[idx]) + add;
  float u = 0.7978845608028654f * (v + 0.044715f * v * v * v);   // tanh GELU
  out[idx] = f2bfu(0.5f * v * (1.f + tanhf(u)));
}

// ---------------- final: y = x1 + fc2out + adapter-up (+expert col scale) ------------
__global__ __launch_bounds__(256) void final_add_kernel(const float* __restrict__ out2,
                                                        const float* __restrict__ t2,
                                                        const float* __restrict__ Bback,
                                                        const float* __restrict__ Bexp,
                                                        const float* __restrict__ expb,
                                                        float* __restrict__ y) {
  int idx = blockIdx.x * 256 + threadIdx.x;     // M_*D_ elements
  int m = idx / D_, n = idx % D_;
  int s = m & (S_ - 1);
  bool expert = (s >= NB_);
  const float* Bm = expert ? Bexp : Bback;
  const float* tr = t2 + (size_t)m * R_;
  float add = 0.f;
#pragma unroll
  for (int r = 0; r < R_; ++r) add += tr[r] * Bm[r * D_ + n];
  if (expert) add *= expb[n];
  y[idx] = y[idx] + out2[idx] + add;
}

// ---------------- host launch ----------------
extern "C" void kernel_launch(void* const* d_in, const int* in_sizes, int n_in,
                              void* d_out, int out_size, void* d_ws, size_t ws_size,
                              hipStream_t stream) {
  (void)in_sizes; (void)n_in; (void)out_size; (void)ws_size;
  const float* x     = (const float*)d_in[0];
  const float* amask = (const float*)d_in[1];
  const float* n1g   = (const float*)d_in[2];
  const float* n1b   = (const float*)d_in[3];
  const float* q_w   = (const float*)d_in[4];
  const float* q_b   = (const float*)d_in[5];
  const float* k_w   = (const float*)d_in[6];
  const float* k_b   = (const float*)d_in[7];
  const float* v_w   = (const float*)d_in[8];
  const float* v_b   = (const float*)d_in[9];
  const float* o_w   = (const float*)d_in[10];
  const float* o_b   = (const float*)d_in[11];
  const float* n2g   = (const float*)d_in[12];
  const float* n2b   = (const float*)d_in[13];
  const float* fc1_w = (const float*)d_in[14];
  const float* fc1_b = (const float*)d_in[15];
  const float* fc2_w = (const float*)d_in[16];
  const float* fc2_b = (const float*)d_in[17];
  const float* lora1A = (const float*)d_in[18];
  const float* lora1B = (const float*)d_in[19];
  const float* lora2A = (const float*)d_in[20];
  const float* lora2B = (const float*)d_in[21];
  const float* vera1A = (const float*)d_in[22];
  const float* vera1B = (const float*)d_in[23];
  const float* vera2A = (const float*)d_in[24];
  const float* vera2B = (const float*)d_in[25];
  const float* exp1d  = (const float*)d_in[26];
  const float* exp1b  = (const float*)d_in[27];
  const float* exp2d  = (const float*)d_in[28];
  const float* exp2b  = (const float*)d_in[29];
  float* y = (float*)d_out;   // also serves as x1 (post-attention residual) buffer

  // workspace carve
  char* p = (char*)d_ws;
  auto carve = [&](size_t bytes) -> char* {
    char* r = p; p += (bytes + 255) & ~(size_t)255; return r;
  };
  const size_t DD = (size_t)D_ * D_, DF = (size_t)D_ * FF_;
  u16* wq   = (u16*)carve(DD * 2);
  u16* wk   = (u16*)carve(DD * 2);
  u16* wv   = (u16*)carve(DD * 2);
  u16* wo   = (u16*)carve(DD * 2);
  u16* wf1  = (u16*)carve(DF * 2);
  u16* wf2  = (u16*)carve(DF * 2);
  u16* hbf  = (u16*)carve((size_t)M_ * D_ * 2);
  u16* qhd  = (u16*)carve((size_t)M_ * D_ * 2);
  u16* khd  = (u16*)carve((size_t)M_ * D_ * 2);
  u16* vhd  = (u16*)carve((size_t)M_ * D_ * 2);
  u16* aob  = (u16*)carve((size_t)M_ * D_ * 2);
  u16* n2bf = (u16*)carve((size_t)M_ * D_ * 2);
  u16* hid1 = (u16*)carve((size_t)M_ * FF_ * 2);
  u16* hact = (u16*)carve((size_t)M_ * FF_ * 2);
  float* out2 = (float*)carve((size_t)M_ * D_ * 4);
  float* t1   = (float*)carve((size_t)M_ * R_ * 4);
  float* t2   = (float*)carve((size_t)M_ * R_ * 4);

  // 1) weights -> bf16
  cvt_bf16_kernel<<<(int)(DD / 256), 256, 0, stream>>>(q_w, wq, (int)DD);
  cvt_bf16_kernel<<<(int)(DD / 256), 256, 0, stream>>>(k_w, wk, (int)DD);
  cvt_bf16_kernel<<<(int)(DD / 256), 256, 0, stream>>>(v_w, wv, (int)DD);
  cvt_bf16_kernel<<<(int)(DD / 256), 256, 0, stream>>>(o_w, wo, (int)DD);
  cvt_bf16_kernel<<<(int)(DF / 256), 256, 0, stream>>>(fc1_w, wf1, (int)DF);
  cvt_bf16_kernel<<<(int)(DF / 256), 256, 0, stream>>>(fc2_w, wf2, (int)DF);

  // 2) LN1
  ln_bf16_kernel<<<M_, 256, 0, stream>>>(x, n1g, n1b, hbf);

  // 3) QKV projections -> head layout bf16 (q pre-scaled by 1/sqrt(64))
  dim3 gD(D_ / 128, M_ / 128);
  gemm_bf16_kernel<<<gD, 256, 0, stream>>>(hbf, wq, q_b, nullptr, nullptr, qhd,
                                           M_, D_, D_, 2, 0.125f);
  gemm_bf16_kernel<<<gD, 256, 0, stream>>>(hbf, wk, k_b, nullptr, nullptr, khd,
                                           M_, D_, D_, 2, 1.0f);
  gemm_bf16_kernel<<<gD, 256, 0, stream>>>(hbf, wv, v_b, nullptr, nullptr, vhd,
                                           M_, D_, D_, 2, 1.0f);

  // 4) flash attention -> merged-head bf16 [B*S, D]
  flash_attn_kernel<<<8 * H_ * (S_ / 128), 256, 0, stream>>>(qhd, khd, vhd, amask, aob);

  // 5) O-proj + residual -> y (x1, f32)
  gemm_bf16_kernel<<<gD, 256, 0, stream>>>(aob, wo, o_b, x, y, nullptr,
                                           M_, D_, D_, 0, 1.0f);

  // 6) LN2
  ln_bf16_kernel<<<M_, 256, 0, stream>>>(y, n2g, n2b, n2bf);

  // 7) FC1 -> bf16 pre-activation
  dim3 gF(FF_ / 128, M_ / 128);
  gemm_bf16_kernel<<<gF, 256, 0, stream>>>(n2bf, wf1, fc1_b, nullptr, nullptr, hid1,
                                           M_, FF_, D_, 1, 1.0f);

  // 8) adapter1 down (LoRA backbone / VeRA expert with diag scale)
  adapter_down_kernel<<<M_, 256, 0, stream>>>(n2bf, lora1A, vera1A, exp1d, t1, D_);

  // 9) adapter1 up + GELU -> bf16
  fc1_post_kernel<<<(M_ * FF_) / 256, 256, 0, stream>>>(hid1, t1, lora1B, vera1B,
                                                        exp1b, hact);

  // 10) FC2 -> f32
  gemm_bf16_kernel<<<gD, 256, 0, stream>>>(hact, wf2, fc2_b, nullptr, out2, nullptr,
                                           M_, D_, FF_, 0, 1.0f);

  // 11) adapter2 down (on post-GELU hidden)
  adapter_down_kernel<<<M_, 256, 0, stream>>>(hact, lora2A, vera2A, exp2d, t2, FF_);

  // 12) final: y = x1 + fc2out + adapter2-up
  final_add_kernel<<<(M_ * D_) / 256, 256, 0, stream>>>(out2, t2, lora2B, vera2B,
                                                        exp2b, y);
}
